// SwinTransformerBlock_72018011619785
// MI455X (gfx1250) — compile-verified, matrix operand order checked by reading, not run on hardware
//
#include <hip/hip_runtime.h>
#include <math.h>

typedef __attribute__((ext_vector_type(16))) _Float16 v16h;
typedef __attribute__((ext_vector_type(8)))  float    v8f;

#define WIN   7
#define NWL   49          // tokens per window
#define NPAD  64          // padded window length
#define CDIM  128
#define NHEAD 4
#define HDIM  32
#define HH    56
#define WWID  56
#define NTOK  (HH*WWID)   // 3136
#define BATCH 32
#define ROWS  (BATCH*NTOK) // 100352
#define SHIFT 3
#define HIDD  512
#define QKVC  384

// ---------------------------------------------------------------------------
// WMMA helpers (gfx1250, wave32). 16-bit A-operand fragment: lane L&15 = row,
// VGPR j holds K = (j<4 ? 2j : 2j+8) + 8*(L>>4), packed pairs (K,K+1).
// B operand (KxN) == A-layout fragment of B^T, loaded via dual strides.
// C/D f32: element i of v8f = D[i + 8*(lane>>4)][lane&15].
// ---------------------------------------------------------------------------
__device__ __forceinline__ v16h load_frag(const _Float16* p, int row_stride,
                                          int k_stride, int lane) {
  int r = lane & 15, hf = lane >> 4;
  v16h f;
#pragma unroll
  for (int j = 0; j < 8; ++j) {
    int kb = ((j < 4) ? (2 * j) : (2 * j + 8)) + 8 * hf;
    f[2 * j]     = p[r * row_stride + kb * k_stride];
    f[2 * j + 1] = p[r * row_stride + (kb + 1) * k_stride];
  }
  return f;
}

__device__ __forceinline__ v8f wmma_f16(v16h a, v16h b, v8f c) {
  return __builtin_amdgcn_wmma_f32_16x16x32_f16(false, a, false, b,
                                                (short)0, c, false, false);
}

// window-order row -> natural token index (shared by LN1 gather & proj scatter;
// roll(-S) before partition and roll(+S) after reverse use the same mapping)
__device__ __forceinline__ long win_row_to_token(int r) {
  int b_ = r / NWL, nn = r % NWL;
  int b = b_ >> 6, win = b_ & 63;
  int hs = (win >> 3) * WIN + nn / WIN;
  int ws = (win & 7) * WIN + nn % WIN;
  int hh = hs + SHIFT; if (hh >= HH)   hh -= HH;
  int ww = ws + SHIFT; if (ww >= WWID) ww -= WWID;
  return (long)b * NTOK + hh * WWID + ww;
}

// ---------------------------------------------------------------------------
// Weight transpose + f32->f16 convert: w[K,N] -> wt[N,K]
// ---------------------------------------------------------------------------
__global__ void k_wt(const float* __restrict__ w, _Float16* __restrict__ wt,
                     int K, int N) {
  int idx = blockIdx.x * 256 + threadIdx.x;
  if (idx >= K * N) return;
  int n = idx / K, k = idx % K;
  wt[idx] = (_Float16)w[(long)k * N + n];
}

// ---------------------------------------------------------------------------
// LayerNorm (one wave per row, C=128 -> 4 floats/lane, xor-shuffle reduce).
// SHIFTED=1: gather through cyclic-shift + window partition (LN1 path).
// ---------------------------------------------------------------------------
template <int SHIFTED>
__global__ __launch_bounds__(256) void k_ln(const float* __restrict__ x,
                                            const float* __restrict__ g,
                                            const float* __restrict__ bta,
                                            _Float16* __restrict__ out) {
  int lane = threadIdx.x & 31;
  int wave = threadIdx.x >> 5;
  int r = blockIdx.x * 8 + wave;
  long src = SHIFTED ? win_row_to_token(r) * CDIM : (long)r * CDIM;

  float v[4];
  float s = 0.f;
#pragma unroll
  for (int u = 0; u < 4; ++u) { v[u] = x[src + lane + 32 * u]; s += v[u]; }
#pragma unroll
  for (int d = 16; d >= 1; d >>= 1) s += __shfl_xor(s, d, 32);
  float mu = s * (1.0f / 128.0f);
  float q = 0.f;
#pragma unroll
  for (int u = 0; u < 4; ++u) { float d0 = v[u] - mu; q += d0 * d0; }
#pragma unroll
  for (int d = 16; d >= 1; d >>= 1) q += __shfl_xor(q, d, 32);
  float inv = rsqrtf(q * (1.0f / 128.0f) + 1e-5f);
#pragma unroll
  for (int u = 0; u < 4; ++u) {
    int c = lane + 32 * u;
    out[(long)r * CDIM + c] = (_Float16)((v[u] - mu) * inv * g[c] + bta[c]);
  }
}

// ---------------------------------------------------------------------------
// WMMA GEMM: out[ROWS,Cout] = A[ROWS,K] @ Wt[Cout,K]^T + bias, fused epilogues.
// Block = 256 threads = 8 waves = 64x32 output tile (each wave one 16x16).
// Activations stay L2-resident (192MB), so direct-global operand fetch is the
// bandwidth-optimal plan on MI455X; no LDS staging needed.
// MODE 0: -> f16            (QKV)
// MODE 1: window->token scatter + residual(x) -> f32   (proj)
// MODE 2: exact GELU -> f16 (fc1)
// MODE 3: + residual(x1) -> f32 (fc2, final output)
// ---------------------------------------------------------------------------
template <int MODE, int K>
__global__ __launch_bounds__(256) void k_gemm(const _Float16* __restrict__ A,
                                              const _Float16* __restrict__ Wt,
                                              const float* __restrict__ bias,
                                              const float* __restrict__ resid,
                                              float* __restrict__ outF,
                                              _Float16* __restrict__ outH,
                                              int Cout) {
  int lane = threadIdx.x & 31;
  int wave = threadIdx.x >> 5;
  int row0 = blockIdx.x * 64 + (wave & 3) * 16;
  int col0 = blockIdx.y * 32 + (wave >> 2) * 16;

  const _Float16* Ap = A  + (long)row0 * K;
  const _Float16* Bp = Wt + (long)col0 * K;
  v8f acc = {};
#pragma unroll
  for (int kt = 0; kt < K; kt += 32) {
    v16h a = load_frag(Ap + kt, K, 1, lane);
    v16h b = load_frag(Bp + kt, K, 1, lane);
    acc = wmma_f16(a, b, acc);
  }

  int n  = col0 + (lane & 15);
  int mb = row0 + ((lane >> 4) << 3);
  float bn = bias[n];
#pragma unroll
  for (int i = 0; i < 8; ++i) {
    int m = mb + i;
    float v = acc[i] + bn;
    if (MODE == 0) {
      outH[(long)m * Cout + n] = (_Float16)v;
    } else if (MODE == 1) {
      long t = win_row_to_token(m) * CDIM + n;
      outF[t] = resid[t] + v;
    } else if (MODE == 2) {
      float ge = 0.5f * v * (1.0f + erff(v * 0.70710678118654752f));
      outH[(long)m * Cout + n] = (_Float16)ge;
    } else {
      long t = (long)m * Cout + n;
      outF[t] = resid[t] + v;
    }
  }
}

// ---------------------------------------------------------------------------
// Windowed attention: one block per window (2048 blocks, 8 waves).
// The window's qkv tile (49x384 f16, one contiguous 36.75KB block) is staged
// raw into LDS via GLOBAL_LOAD_ASYNC_TO_LDS_B128 (ASYNCcnt path), rows 49..63
// zero-padded. Fragment loaders index the raw tile directly:
//   Q[h]:  sRaw[n][       h*32 + d]   (row stride 384)
//   K[h]:  sRaw[m][128 + h*32 + d]   (row stride 384, used as B = K^T)
//   V[h]:  sRaw[k][256 + h*32 + c]   (as B: strides (1, 384))
// Wave task = (head, 16-row tile): S = Q K^T (4 WMMA), fused rel-pos bias +
// shift mask + softmax, then out = P V (4 WMMA) -> f16 [ROWS,128].
// ---------------------------------------------------------------------------
__global__ __launch_bounds__(256) void k_attn(const _Float16* __restrict__ qkv,
                                              const float* __restrict__ rbt,
                                              const float* __restrict__ amask,
                                              _Float16* __restrict__ out) {
  __shared__ _Float16 sRaw[NPAD][QKVC];   // 48KB: window qkv tile, padded rows
  __shared__ _Float16 sP[8][16][NPAD];    // 16KB: per-wave P staging

  int tid = threadIdx.x;
  int win = blockIdx.x;
  int mw  = win & 63;  // window index within image (mask row)

  // zero the padding rows (49..63) with regular DS stores
  for (int idx = tid; idx < (NPAD - NWL) * QKVC; idx += 256)
    sRaw[NWL][0 + idx] = (_Float16)0.f;

  // async-copy the 49x384 f16 tile (contiguous 37632 bytes) into LDS
  {
    const char* gbase = (const char*)(qkv + (long)win * NWL * QKVC);
    unsigned lds0 = (unsigned)(size_t)(&sRaw[0][0]);  // LDS window offset
    const int total = NWL * QKVC * 2;                 // 37632 bytes
    for (int byte = tid * 16; byte < total; byte += 256 * 16) {
      unsigned ldsa = lds0 + (unsigned)byte;
      unsigned voff = (unsigned)byte;
      asm volatile("global_load_async_to_lds_b128 %0, %1, %2"
                   :
                   : "v"(ldsa), "v"(voff), "s"(gbase)
                   : "memory");
    }
    asm volatile("s_wait_asynccnt 0x0" ::: "memory");
  }
  __syncthreads();

  int lane = tid & 31, wave = tid >> 5;
  int half = lane >> 4, lc = lane & 15;
  const float scale = 0.17677669529663689f;  // HD^-0.5

  for (int task = wave; task < 16; task += 8) {
    int h = task >> 2, rt = task & 3;
    int qr0 = rt * 16;

    v16h aq = load_frag(&sRaw[qr0][h * HDIM], QKVC, 1, lane);
    v8f sc[4];
#pragma unroll
    for (int ct = 0; ct < 4; ++ct) {
      v8f z = {};
      v16h bk = load_frag(&sRaw[ct * 16][CDIM + h * HDIM], QKVC, 1, lane);
      sc[ct] = wmma_f16(aq, bk, z);
    }

    // fused scale + rel-pos bias + shift mask, then row softmax.
    float p[4][8];
#pragma unroll
    for (int i = 0; i < 8; ++i) {
      int mr = qr0 + i + 8 * half;                 // query row in [0,64)
      float mx = -1e30f;
#pragma unroll
      for (int ct = 0; ct < 4; ++ct) {
        int kc = ct * 16 + lc;                     // key column in [0,64)
        float v;
        if (mr < NWL && kc < NWL) {
          int i1 = mr / WIN, j1 = mr % WIN, i2 = kc / WIN, j2 = kc % WIN;
          int ri = (i1 - i2 + WIN - 1) * (2 * WIN - 1) + (j1 - j2 + WIN - 1);
          v = sc[ct][i] * scale + rbt[ri * NHEAD + h]
              + amask[((long)mw * NWL + mr) * NWL + kc];
        } else {
          v = -1e30f;  // padded keys excluded from softmax
        }
        p[ct][i] = v;
        mx = fmaxf(mx, v);
      }
      // a row lives in one 16-lane half: xor masks 1..8 stay inside it.
#pragma unroll
      for (int d = 1; d <= 8; d <<= 1) mx = fmaxf(mx, __shfl_xor(mx, d, 32));
      float s = 0.f;
#pragma unroll
      for (int ct = 0; ct < 4; ++ct) { p[ct][i] = __expf(p[ct][i] - mx); s += p[ct][i]; }
#pragma unroll
      for (int d = 1; d <= 8; d <<= 1) s += __shfl_xor(s, d, 32);
      float inv = 1.0f / s;
#pragma unroll
      for (int ct = 0; ct < 4; ++ct) p[ct][i] *= inv;
    }

    // C-layout -> A-layout through per-wave LDS staging
#pragma unroll
    for (int i = 0; i < 8; ++i) {
      int rl = i + 8 * half;
#pragma unroll
      for (int ct = 0; ct < 4; ++ct)
        sP[wave][rl][ct * 16 + lc] = (_Float16)p[ct][i];
    }
    __syncthreads();

    v8f o[2] = {};
#pragma unroll
    for (int kt = 0; kt < 2; ++kt) {
      v16h ap = load_frag(&sP[wave][0][kt * 32], NPAD, 1, lane);
#pragma unroll
      for (int ct2 = 0; ct2 < 2; ++ct2) {
        // B = V[k][c]: element (c,k) = sRaw[kt*32+k][256 + h*32 + ct2*16 + c]
        v16h bv = load_frag(&sRaw[kt * 32][2 * CDIM + h * HDIM + ct2 * 16],
                            1, QKVC, lane);
        o[ct2] = wmma_f16(ap, bv, o[ct2]);
      }
    }
#pragma unroll
    for (int ct2 = 0; ct2 < 2; ++ct2) {
#pragma unroll
      for (int i = 0; i < 8; ++i) {
        int qr = qr0 + i + 8 * half;
        if (qr < NWL) {
          int c = h * HDIM + ct2 * 16 + lc;
          out[((long)win * NWL + qr) * CDIM + c] = (_Float16)o[ct2][i];
        }
      }
    }
    __syncthreads();
  }
}

// ---------------------------------------------------------------------------
extern "C" void kernel_launch(void* const* d_in, const int* in_sizes, int n_in,
                              void* d_out, int out_size, void* d_ws,
                              size_t ws_size, hipStream_t stream) {
  const float* x      = (const float*)d_in[0];
  const float* amask  = (const float*)d_in[3];
  const float* n1g    = (const float*)d_in[4];
  const float* n1b    = (const float*)d_in[5];
  const float* qkv_w  = (const float*)d_in[6];
  const float* qkv_b  = (const float*)d_in[7];
  const float* rbt    = (const float*)d_in[8];
  const float* proj_w = (const float*)d_in[9];
  const float* proj_b = (const float*)d_in[10];
  const float* n2g    = (const float*)d_in[11];
  const float* n2b    = (const float*)d_in[12];
  const float* fc1_w  = (const float*)d_in[13];
  const float* fc1_b  = (const float*)d_in[14];
  const float* fc2_w  = (const float*)d_in[15];
  const float* fc2_b  = (const float*)d_in[16];
  float* outp = (float*)d_out;

  // workspace layout (lifetimes disjoint where aliased): ~180 MB
  char* ws = (char*)d_ws;
  size_t off = 0;
  auto alloc = [&](size_t bytes) {
    char* p = ws + off;
    off += (bytes + 255) & ~(size_t)255;
    return p;
  };
  _Float16* wt_qkv  = (_Float16*)alloc((size_t)QKVC * 128 * 2);
  _Float16* wt_proj = (_Float16*)alloc((size_t)128 * 128 * 2);
  _Float16* wt_fc1  = (_Float16*)alloc((size_t)512 * 128 * 2);
  _Float16* wt_fc2  = (_Float16*)alloc((size_t)128 * 512 * 2);
  char* p1 = alloc((size_t)ROWS * 128 * 2);  // xw -> attn_out -> xn2
  char* p2 = alloc((size_t)ROWS * 512 * 2);  // qkv -> h
  float* x1 = (float*)alloc((size_t)ROWS * 128 * 4);

  _Float16* xw       = (_Float16*)p1;
  _Float16* qkvbuf   = (_Float16*)p2;
  _Float16* attn_out = (_Float16*)p1;
  _Float16* xn2      = (_Float16*)p1;
  _Float16* hbuf     = (_Float16*)p2;

  // weight prep
  k_wt<<<(QKVC * 128 + 255) / 256, 256, 0, stream>>>(qkv_w, wt_qkv, 128, QKVC);
  k_wt<<<(128 * 128 + 255) / 256, 256, 0, stream>>>(proj_w, wt_proj, 128, 128);
  k_wt<<<(512 * 128 + 255) / 256, 256, 0, stream>>>(fc1_w, wt_fc1, 128, 512);
  k_wt<<<(128 * 512 + 255) / 256, 256, 0, stream>>>(fc2_w, wt_fc2, 512, 128);

  // LN1 + shift + window partition
  k_ln<1><<<ROWS / 8, 256, 0, stream>>>(x, n1g, n1b, xw);
  // QKV
  {
    dim3 g(ROWS / 64, QKVC / 32);
    k_gemm<0, 128><<<g, 256, 0, stream>>>(xw, wt_qkv, qkv_b, nullptr, nullptr,
                                          qkvbuf, QKVC);
  }
  // attention
  k_attn<<<BATCH * 64, 256, 0, stream>>>(qkvbuf, rbt, amask, attn_out);
  // proj + window reverse + unshift + residual
  {
    dim3 g(ROWS / 64, 128 / 32);
    k_gemm<1, 128><<<g, 256, 0, stream>>>(attn_out, wt_proj, proj_b, x, x1,
                                          nullptr, 128);
  }
  // LN2
  k_ln<0><<<ROWS / 8, 256, 0, stream>>>(x1, n2g, n2b, xn2);
  // fc1 + GELU
  {
    dim3 g(ROWS / 64, 512 / 32);
    k_gemm<2, 128><<<g, 256, 0, stream>>>(xn2, wt_fc1, fc1_b, nullptr, nullptr,
                                          hbuf, 512);
  }
  // fc2 + residual -> final output
  {
    dim3 g(ROWS / 64, 128 / 32);
    k_gemm<3, 512><<<g, 256, 0, stream>>>(hbuf, wt_fc2, fc2_b, x1, outp,
                                          nullptr, 128);
  }
}
